// WaveFunctionElectron_44513041056433
// MI455X (gfx1250) — compile-verified, matrix operand order checked by reading, not run on hardware
//
#include <hip/hip_runtime.h>
#include <hip/hip_bf16.h>

// ---------------------------------------------------------------------------
// Problem constants (match reference)
// ---------------------------------------------------------------------------
#define S_ 128    // sites
#define B_ 2048   // batch
#define H_ 256    // hidden
#define MT 32     // batch rows per block (site kernel)
#define XSTR 136  // LDS stride for x tile  (272B = 17*16B)
#define HSTR 264  // LDS stride for h tiles (528B = 33*16B)

typedef __bf16 bf16;
typedef __attribute__((ext_vector_type(16))) __bf16 v16bf;
typedef __attribute__((ext_vector_type(8)))  __bf16 v8bf;
typedef __attribute__((ext_vector_type(8)))  float  v8f;

// ---------------------------------------------------------------------------
// WMMA helpers
// ---------------------------------------------------------------------------
__device__ __forceinline__ v8f wmma_bf(v16bf a, v16bf b, v8f c) {
  // D = A(16x32 bf16) * B(32x16 bf16) + C(16x16 f32)
  return __builtin_amdgcn_wmma_f32_16x16x32_bf16(
      /*neg_a=*/false, a, /*neg_b=*/false, b,
      /*c_mod=*/(short)0, c, /*reuse_a=*/false, /*reuse_b=*/false);
}

// 16-element bf16 fragment: two contiguous 8-elem (16B) runs at p[0..7] and
// p[16..23] (ISA 16-bit fragment layout: lane's K = {base..+7} U {base+16..+23})
__device__ __forceinline__ v16bf frag16(const bf16* p) {
  v8bf lo = *(const v8bf*)(p);
  v8bf hi = *(const v8bf*)(p + 16);
  v16bf r;
#pragma unroll
  for (int j = 0; j < 8; ++j) { r[j] = lo[j]; r[j + 8] = hi[j]; }
  return r;
}

__device__ __forceinline__ float silu(float v) {
  return v / (1.0f + __expf(-v));
}

// Pipeline stage: 2 A fragments (rows 0-15 / 16-31 of the 32-row tile) and
// 2 B fragments (the wave's 32 weight columns).
struct Frags { v16bf a0, a1, b0, b1; };

__device__ __forceinline__ Frags load_frags(const bf16* lh, int lstride,
                                            const bf16* w0, const bf16* w1,
                                            int arow, int kb) {
  Frags f;
  f.a0 = frag16(lh + arow * lstride + kb);
  f.a1 = frag16(lh + (arow + 16) * lstride + kb);
  f.b0 = frag16(w0 + kb);
  f.b1 = frag16(w1 + kb);
  return f;
}

__device__ __forceinline__ void mma4(const Frags& f, v8f (&acc)[2][2]) {
  acc[0][0] = wmma_bf(f.a0, f.b0, acc[0][0]);
  acc[0][1] = wmma_bf(f.a0, f.b1, acc[0][1]);
  acc[1][0] = wmma_bf(f.a1, f.b0, acc[1][0]);
  acc[1][1] = wmma_bf(f.a1, f.b1, acc[1][1]);
}

// ---------------------------------------------------------------------------
// 1) Convert + transpose weights to bf16 (one-time; then L2-resident)
//    W1t[i][h][s] = W1[i][s][h]   (bf16)   -> B-fragment K-runs contiguous
//    W2t[i][k2][k1] = W2[i][k1][k2] (bf16)
// ---------------------------------------------------------------------------
__global__ __launch_bounds__(256) void prep_weights(
    const float* __restrict__ W1, const float* __restrict__ W2,
    bf16* __restrict__ W1t, bf16* __restrict__ W2t) {
  const size_t N1 = (size_t)S_ * H_ * S_;      // 4,194,304
  const size_t N2 = (size_t)S_ * H_ * H_;      // 8,388,608
  size_t idx = (size_t)blockIdx.x * blockDim.x + threadIdx.x;
  size_t stride = (size_t)gridDim.x * blockDim.x;
  for (; idx < N1 + N2; idx += stride) {
    if (idx < N1) {
      size_t i = idx / ((size_t)H_ * S_);
      size_t rem = idx % ((size_t)H_ * S_);
      size_t h = rem / S_, s = rem % S_;
      W1t[idx] = (bf16)W1[(i * S_ + s) * H_ + h];
    } else {
      size_t j = idx - N1;
      size_t i = j / ((size_t)H_ * H_);
      size_t rem = j % ((size_t)H_ * H_);
      size_t k2 = rem / H_, k1 = rem % H_;
      W2t[j] = (bf16)W2[(i * H_ + k1) * H_ + k2];
    }
  }
}

// ---------------------------------------------------------------------------
// 2) Exclusive cumsum per row (electrons before site i) + zero amp accumulator
// ---------------------------------------------------------------------------
__global__ __launch_bounds__(256) void prep_counts(
    const int* __restrict__ x, int* __restrict__ counts,
    float* __restrict__ amp) {
  int b = blockIdx.x * blockDim.x + threadIdx.x;
  if (b >= B_) return;
  int c = 0;
  for (int s = 0; s < S_; ++s) { counts[b * S_ + s] = c; c += x[b * S_ + s]; }
  amp[b] = 0.0f;
}

// ---------------------------------------------------------------------------
// 3) Per-site fused MLP: grid = (site i, batch tile of MT rows), 8 waves.
//    Each wave owns a 32x32 output slab (2 M-tiles x 2 N-tiles) so every
//    weight element is fetched exactly once per block; K loops are fully
//    unrolled with register ping-pong so WMMAs overlap L2 latency.
// ---------------------------------------------------------------------------
__global__ __launch_bounds__(256) void site_kernel(
    const int* __restrict__ x, const int* __restrict__ counts,
    const bf16* __restrict__ W1t, const bf16* __restrict__ W2t,
    const float* __restrict__ b1, const float* __restrict__ b2,
    const float* __restrict__ W3, const float* __restrict__ b3,
    const int* __restrict__ eptr, float* __restrict__ amp) {
  __shared__ __align__(16) bf16 lx[MT * XSTR];     //  8,704 B
  __shared__ __align__(16) bf16 lh1[MT * HSTR];    // 16,896 B
  __shared__ __align__(16) bf16 lh2[MT * HSTR];    // 16,896 B
  __shared__ float llog[MT][2];

  const int i = blockIdx.x;           // site
  const int b0 = blockIdx.y * MT;     // batch tile base
  const int tid = threadIdx.x;
  const int lane = tid & 31;
  const int w = tid >> 5;
  const int E = eptr[0];

  // Stage x tile with causal mask applied (s < i, else 0)
  for (int t = tid; t < MT * S_; t += 256) {
    int m = t >> 7, s = t & (S_ - 1);
    float v = (s < i) ? (float)x[(b0 + m) * S_ + s] : 0.0f;
    lx[m * XSTR + s] = (bf16)v;
  }
  __syncthreads();

  const int nb = w * 32;                         // wave's 32 weight columns
  const int arow = lane & 15;                    // A fragment row within tile
  const int khalf = (lane >> 4) * 8;             // fragment K sub-offset
  const int half8 = (lane >> 4) << 3;            // C row sub-base
  const int ncol = lane & 15;

  v8f acc[2][2];
#pragma unroll
  for (int a = 0; a < 2; ++a)
#pragma unroll
    for (int bcol = 0; bcol < 2; ++bcol)
#pragma unroll
      for (int r = 0; r < 8; ++r) acc[a][bcol][r] = 0.0f;

  // ---- Layer 1: [MTx128 causal-masked] x [128x256] ----
  const bf16* W1r0 = W1t + (size_t)i * H_ * S_ + (size_t)(nb + ncol) * S_;
  const bf16* W1r1 = W1r0 + (size_t)16 * S_;
  const bf16* W2r0 = W2t + (size_t)i * H_ * H_ + (size_t)(nb + ncol) * H_;
  const bf16* W2r1 = W2r0 + (size_t)16 * H_;
  // warm layer-2 weights while layer-1 runs
  __builtin_prefetch(W2r0, 0, 1);
  __builtin_prefetch(W2r1, 0, 1);
  {
    Frags f = load_frags(lx, XSTR, W1r0, W1r1, arow, khalf);
#pragma unroll
    for (int kt = 0; kt < 4; ++kt) {
      Frags nf = f;
      if (kt < 3) nf = load_frags(lx, XSTR, W1r0, W1r1, arow, (kt + 1) * 32 + khalf);
      mma4(f, acc);
      f = nf;
    }
  }
#pragma unroll
  for (int a = 0; a < 2; ++a)
#pragma unroll
    for (int bcol = 0; bcol < 2; ++bcol) {
      int n = nb + bcol * 16 + ncol;
      float bias = b1[i * H_ + n];
#pragma unroll
      for (int r = 0; r < 8; ++r) {
        float v = silu(acc[a][bcol][r] + bias);
        lh1[(a * 16 + half8 + r) * HSTR + n] = (bf16)v;
        acc[a][bcol][r] = 0.0f;
      }
    }
  __syncthreads();

  // ---- Layer 2: [MTx256] x [256x256] ----
  {
    Frags f = load_frags(lh1, HSTR, W2r0, W2r1, arow, khalf);
#pragma unroll
    for (int kt = 0; kt < 8; ++kt) {
      Frags nf = f;
      if (kt < 7) nf = load_frags(lh1, HSTR, W2r0, W2r1, arow, (kt + 1) * 32 + khalf);
      mma4(f, acc);
      f = nf;
    }
  }
#pragma unroll
  for (int a = 0; a < 2; ++a)
#pragma unroll
    for (int bcol = 0; bcol < 2; ++bcol) {
      int n = nb + bcol * 16 + ncol;
      float bias = b2[i * H_ + n];
#pragma unroll
      for (int r = 0; r < 8; ++r) {
        float v = silu(acc[a][bcol][r] + bias);
        lh2[(a * 16 + half8 + r) * HSTR + n] = (bf16)v;
      }
    }
  __syncthreads();

  // ---- Layer 3 (Hx2) + mask + log-norm + select + accumulate ----
  if (tid < MT * 2) {
    int m = tid >> 1, o = tid & 1;
    float s = b3[i * 2 + o];
    const float* w3 = W3 + (size_t)i * H_ * 2 + o;
    for (int k = 0; k < H_; ++k) s += (float)lh2[m * HSTR + k] * w3[k * 2];
    llog[m][o] = s;
  }
  __syncthreads();
  if (tid < MT) {
    int b = b0 + tid;
    int c = counts[b * S_ + i];
    bool mh = (i - c) < (S_ - E);    // may add hole (o=0)
    bool me = c < E;                  // may add electron (o=1)
    float l0 = llog[tid][0], l1 = llog[tid][1];
    float v0 = mh ? l0 : -3.0e38f;
    float v1 = me ? l1 : -3.0e38f;
    float mx = fmaxf(v0, v1);
    float z = (mh ? __expf(2.0f * (v0 - mx)) : 0.0f) +
              (me ? __expf(2.0f * (v1 - mx)) : 0.0f);
    float lse = mx + 0.5f * __logf(z);
    int xb = x[b * S_ + i];
    float sel = (xb ? l1 : l0) - lse;
    atomicAdd(&amp[b], sel);
  }
}

// ---------------------------------------------------------------------------
// 4) Phase MLP (tiny) + complex combine: one block per config
// ---------------------------------------------------------------------------
__global__ __launch_bounds__(256) void phase_kernel(
    const int* __restrict__ x,
    const float* __restrict__ Wp1, const float* __restrict__ bp1,
    const float* __restrict__ Wp2, const float* __restrict__ bp2,
    const float* __restrict__ Wp3, const float* __restrict__ bp3,
    const float* __restrict__ amp, float* __restrict__ out) {
  __shared__ float xs[S_];
  __shared__ float pb[H_];
  __shared__ float qb[H_];
  __shared__ float red[256];
  int b = blockIdx.x, t = threadIdx.x;
  if (t < S_) xs[t] = (float)x[b * S_ + t];
  __syncthreads();
  float a = bp1[t];
  for (int s = 0; s < S_; ++s) a += xs[s] * Wp1[s * H_ + t];
  pb[t] = silu(a);
  __syncthreads();
  a = bp2[t];
  for (int k = 0; k < H_; ++k) a += pb[k] * Wp2[k * H_ + t];
  qb[t] = silu(a);
  __syncthreads();
  red[t] = qb[t] * Wp3[t];
  __syncthreads();
  for (int off = 128; off > 0; off >>= 1) {
    if (t < off) red[t] += red[t + off];
    __syncthreads();
  }
  if (t == 0) {
    float ph = red[0] + bp3[0];
    float ea = __expf(amp[b]);
    out[2 * b + 0] = ea * __cosf(ph);   // complex64: interleaved re, im
    out[2 * b + 1] = ea * __sinf(ph);
  }
}

// ---------------------------------------------------------------------------
// Launch
// ---------------------------------------------------------------------------
extern "C" void kernel_launch(void* const* d_in, const int* in_sizes, int n_in,
                              void* d_out, int out_size, void* d_ws, size_t ws_size,
                              hipStream_t stream) {
  const int*   x    = (const int*)d_in[0];
  const int*   eptr = (const int*)d_in[1];
  const float* W1   = (const float*)d_in[2];
  const float* b1   = (const float*)d_in[3];
  const float* W2   = (const float*)d_in[4];
  const float* b2   = (const float*)d_in[5];
  const float* W3   = (const float*)d_in[6];
  const float* b3   = (const float*)d_in[7];
  const float* Wp1  = (const float*)d_in[8];
  const float* bp1  = (const float*)d_in[9];
  const float* Wp2  = (const float*)d_in[10];
  const float* bp2  = (const float*)d_in[11];
  const float* Wp3  = (const float*)d_in[12];
  const float* bp3  = (const float*)d_in[13];

  // Workspace layout
  char* ws = (char*)d_ws;
  bf16*  W1t    = (bf16*)(ws);                          //  8,388,608 B
  bf16*  W2t    = (bf16*)(ws + 8388608);                // 16,777,216 B
  int*   counts = (int*)(ws + 25165824);                //  1,048,576 B
  float* amp    = (float*)(ws + 26214400);              //      8,192 B

  const size_t NW = (size_t)S_ * H_ * S_ + (size_t)S_ * H_ * H_;  // 12,582,912
  prep_weights<<<(int)((NW + 255) / 256), 256, 0, stream>>>(W1, W2, W1t, W2t);
  prep_counts<<<(B_ + 255) / 256, 256, 0, stream>>>(x, counts, amp);
  site_kernel<<<dim3(S_, B_ / MT), 256, 0, stream>>>(
      x, counts, W1t, W2t, b1, b2, W3, b3, eptr, amp);
  phase_kernel<<<B_, 256, 0, stream>>>(x, Wp1, bp1, Wp2, bp2, Wp3, bp3, amp,
                                       (float*)d_out);
}
